// TripletLoss_40218073760104
// MI455X (gfx1250) — compile-verified, matrix operand order checked by reading, not run on hardware
//
#include <hip/hip_runtime.h>

typedef __attribute__((ext_vector_type(16))) _Float16 v16h;
typedef __attribute__((ext_vector_type(8)))  _Float16 h8;
typedef __attribute__((ext_vector_type(2)))  _Float16 h2;
typedef __attribute__((ext_vector_type(8)))  float    v8f;

#define F_DIM   300            // feature dim (K)
#define D_DIM   1000           // classes (N)
#define K_PAD   320            // 10 k-steps of 32 (f16 WMMA depth)
#define N_TILES 63             // ceil(1000/16)
#define M_TILE  32             // rows per block: 2 WMMA M-subtiles -> 2x B reuse
#define S_PITCH 1008           // LDS row pitch for S slab
#define MARGIN  0.1f

#define LDS_A_HALFS (M_TILE * K_PAD)                          // f16 A tile (20 KB)
#define LDS_BYTES   (LDS_A_HALFS * 2 + M_TILE * S_PITCH * 4)  // 149504 B (<320 KB/WG)

#define YH_OFF_BYTES 64                                   // accumulators live at ws[0..1]
#define YH_BYTES     ((size_t)D_DIM * K_PAD * 2)          // 640 KB f16 Y image

__global__ void triplet_init(float* ws) {
    ws[0] = 0.0f;                    // running hinge total (f32)
    ((unsigned int*)ws)[1] = 0u;     // running include count (u32)
}

// One-time Y f32 -> f16 conversion, zero-padded K 300 -> 320.
__global__ void triplet_yconv(const float* __restrict__ Y, _Float16* __restrict__ Yh) {
    const int idx = blockIdx.x * blockDim.x + threadIdx.x;   // half2 index
    const int total = D_DIM * (K_PAD / 2);
    if (idx < total) {
        const int nrow = idx / (K_PAD / 2);
        const int kp   = (idx - nrow * (K_PAD / 2)) * 2;
        const float a = (kp     < F_DIM) ? Y[(size_t)nrow * F_DIM + kp]     : 0.0f;
        const float b = (kp + 1 < F_DIM) ? Y[(size_t)nrow * F_DIM + kp + 1] : 0.0f;
        h2 p; p.x = (_Float16)a; p.y = (_Float16)b;
        *(h2*)(Yh + (size_t)nrow * K_PAD + kp) = p;
    }
}

template <bool PRECONV>
__global__ __launch_bounds__(256) void triplet_fused(
    const float* __restrict__ X, const float* __restrict__ Y,
    const _Float16* __restrict__ Yh,
    const int* __restrict__ yidx, float* __restrict__ ws, int n) {
    extern __shared__ char smem[];
    _Float16* Af = (_Float16*)smem;                   // [M_TILE][K_PAD] f16
    float*    S  = (float*)(smem + LDS_A_HALFS * 2);  // [M_TILE][S_PITCH] f32

    const int tid     = threadIdx.x;
    const int lane    = tid & 31;
    const int wave    = tid >> 5;
    const int rowBase = blockIdx.x * M_TILE;

    // ---- Stage A tile: 32 rows x 300 f32 -> f16 in LDS, zero-pad K to 320 ----
    // Zeros for k>=300 and rows>=n make every OOB contribution exactly 0.
    for (int idx = tid; idx < M_TILE * K_PAD; idx += 256) {
        const int m = idx / K_PAD, k = idx - m * K_PAD;
        const int row = rowBase + m;
        float v = 0.0f;
        if (k < F_DIM && row < n) v = X[(size_t)row * F_DIM + k];
        Af[idx] = (_Float16)v;
    }
    __syncthreads();

    // ---- Preload A fragments for BOTH M-subtiles (invariant across N) ----
    // 16-bit fragment layout (wave32): lane holds M=lane&15; lanes 16-31 get
    // the +8 K swap; each lane's 16 halves are two contiguous 16B runs.
    const int mA  = lane & 15;
    const int hiK = (lane >= 16) ? 8 : 0;
    v16h afrag0[10], afrag1[10];                 // rows 0-15 / rows 16-31
#pragma unroll
    for (int ks = 0; ks < 10; ++ks) {
        const _Float16* a0 = Af + mA * K_PAD + ks * 32 + hiK;
        const _Float16* a1 = a0 + 16 * K_PAD;
        const h8 lo0 = *(const h8*)(a0), hi0 = *(const h8*)(a0 + 16);  // ds_load_b128 x2
        const h8 lo1 = *(const h8*)(a1), hi1 = *(const h8*)(a1 + 16);
        afrag0[ks] = __builtin_shufflevector(lo0, hi0,
            0, 1, 2, 3, 4, 5, 6, 7, 8, 9, 10, 11, 12, 13, 14, 15);
        afrag1[ks] = __builtin_shufflevector(lo1, hi1,
            0, 1, 2, 3, 4, 5, 6, 7, 8, 9, 10, 11, 12, 13, 14, 15);
    }

    // ---- WMMA GEMM over N tiles: each B fragment feeds TWO WMMAs ----
    for (int nt = wave; nt < N_TILES; nt += 8) {
        const int  ncol    = nt * 16 + (lane & 15);
        const bool ncol_ok = (ncol < D_DIM);
        const int  nclamp  = ncol_ok ? ncol : (D_DIM - 1);   // N tail: clamp row

        v8f c0 = {0.f, 0.f, 0.f, 0.f, 0.f, 0.f, 0.f, 0.f};
        v8f c1 = {0.f, 0.f, 0.f, 0.f, 0.f, 0.f, 0.f, 0.f};
        if (PRECONV) {
            // Fast path: B already f16 + zero-padded; 2 aligned b128 loads
            // per fragment, no converts, no clamps, 2 WMMAs per fragment.
            const _Float16* Yrow = Yh + (size_t)nclamp * K_PAD;
#pragma unroll
            for (int ks = 0; ks < 10; ++ks) {
                const _Float16* bp = Yrow + ks * 32 + hiK;
                const h8 lo = *(const h8*)(bp);        // global_load_b128
                const h8 hi = *(const h8*)(bp + 16);   // global_load_b128
                const v16h b = __builtin_shufflevector(lo, hi,
                    0, 1, 2, 3, 4, 5, 6, 7, 8, 9, 10, 11, 12, 13, 14, 15);
                c0 = __builtin_amdgcn_wmma_f32_16x16x32_f16(
                        false, afrag0[ks], false, b, (short)0, c0, false, false);
                c1 = __builtin_amdgcn_wmma_f32_16x16x32_f16(
                        false, afrag1[ks], false, b, (short)0, c1, false, false);
            }
        } else {
            // Fallback: convert f32 Y on the fly, address-clamped (branchless).
            const float* Yrow = Y + (size_t)nclamp * F_DIM;
#pragma unroll
            for (int ks = 0; ks < 10; ++ks) {
                const int kb = ks * 32 + hiK;
                int k0 = kb;      k0 = (k0 <= F_DIM - 4) ? k0 : 0;
                int k1 = kb + 4;  k1 = (k1 <= F_DIM - 4) ? k1 : 0;
                int k2 = kb + 16; k2 = (k2 <= F_DIM - 4) ? k2 : 0;
                int k3 = kb + 20; k3 = (k3 <= F_DIM - 4) ? k3 : 0;
                const float4 f0 = *(const float4*)(Yrow + k0);
                const float4 f1 = *(const float4*)(Yrow + k1);
                const float4 f2 = *(const float4*)(Yrow + k2);
                const float4 f3 = *(const float4*)(Yrow + k3);
                v16h b;
                b[0]  = (_Float16)f0.x; b[1]  = (_Float16)f0.y;
                b[2]  = (_Float16)f0.z; b[3]  = (_Float16)f0.w;
                b[4]  = (_Float16)f1.x; b[5]  = (_Float16)f1.y;
                b[6]  = (_Float16)f1.z; b[7]  = (_Float16)f1.w;
                b[8]  = (_Float16)f2.x; b[9]  = (_Float16)f2.y;
                b[10] = (_Float16)f2.z; b[11] = (_Float16)f2.w;
                b[12] = (_Float16)f3.x; b[13] = (_Float16)f3.y;
                b[14] = (_Float16)f3.z; b[15] = (_Float16)f3.w;
                c0 = __builtin_amdgcn_wmma_f32_16x16x32_f16(
                        false, afrag0[ks], false, b, (short)0, c0, false, false);
                c1 = __builtin_amdgcn_wmma_f32_16x16x32_f16(
                        false, afrag1[ks], false, b, (short)0, c1, false, false);
            }
        }
        // C/D layout: VGPR r, lanes 0-15 -> M=r, lanes 16-31 -> M=r+8; N=lane&15
#pragma unroll
        for (int r = 0; r < 8; ++r) {
            const int m = r + ((lane >= 16) ? 8 : 0);
            if (ncol_ok) {
                S[m * S_PITCH + ncol]        = c0[r];
                S[(m + 16) * S_PITCH + ncol] = c1[r];
            }
        }
    }
    __syncthreads();

    // ---- Loss walk: one lane per row, early break on (ap - an) > margin ----
    if (tid < M_TILE) {
        const int row = rowBase + tid;
        if (row < n) {
            const int   y  = yidx[row];
            const float ap = S[tid * S_PITCH + y];
            // deterministic per-row affine permutation, stride coprime to 1000
            unsigned int h = (unsigned int)row * 2654435761u;
            h ^= h >> 16; h *= 2246822519u; h ^= h >> 13;
            const int stride = 3 + 10 * (int)((h >> 8) % 100u); // =3 mod 10
            int j = (int)(h % (unsigned int)D_DIM);
            float sum = 0.0f; unsigned int cnt = 0u;
            for (int t = 0; t < D_DIM; ++t) {
                if (j != y) {
                    const float an   = S[tid * S_PITCH + j];
                    const float diff = ap - an;
                    sum += fmaxf(0.0f, MARGIN - diff);  // max(0, margin+an-ap)
                    ++cnt;
                    if (diff > MARGIN) break;           // include break slot, stop
                }
                j += stride; if (j >= D_DIM) j -= D_DIM;
            }
            atomicAdd(ws, sum);
            atomicAdd((unsigned int*)ws + 1, cnt);
        }
    }
}

__global__ void triplet_final(const float* __restrict__ ws, float* __restrict__ out) {
    const unsigned int cnt = ((const unsigned int*)ws)[1];
    out[0] = ws[0] / (float)(cnt ? cnt : 1u);
}

extern "C" void kernel_launch(void* const* d_in, const int* in_sizes, int n_in,
                              void* d_out, int out_size, void* d_ws, size_t ws_size,
                              hipStream_t stream) {
    const float* X  = (const float*)d_in[0];
    const float* Y  = (const float*)d_in[1];
    const int*   yi = (const int*)d_in[2];
    const int    n  = in_sizes[2];          // rows = len(y_idxs)
    float* ws = (float*)d_ws;               // [f32 total][u32 count][pad][Yh f16...]

    triplet_init<<<1, 1, 0, stream>>>(ws);
    const int blocks = (n + M_TILE - 1) / M_TILE;

    if (ws_size >= YH_OFF_BYTES + YH_BYTES) {
        _Float16* Yh = (_Float16*)((char*)d_ws + YH_OFF_BYTES);
        const int pairs = D_DIM * (K_PAD / 2);
        triplet_yconv<<<(pairs + 255) / 256, 256, 0, stream>>>(Y, Yh);
        triplet_fused<true><<<blocks, 256, LDS_BYTES, stream>>>(X, Y, Yh, yi, ws, n);
    } else {
        triplet_fused<false><<<blocks, 256, LDS_BYTES, stream>>>(X, Y, nullptr, yi, ws, n);
    }
    triplet_final<<<1, 1, 0, stream>>>(ws, (float*)d_out);
}
